// PaperGNN_12730283065832
// MI455X (gfx1250) — matrix operand order repaired and checked by reading, not compiled, not run on hardware
//
#include <hip/hip_runtime.h>

// ---------------------------------------------------------------------------
// HGT forward for MI455X (gfx1250, wave32, WMMA).
// Dense GEMMs -> v_wmma_f32_16x16x32_bf16; A via unconditional global_load_b128
// (row-clamped, store-guarded); B slab copied global->LDS with
// global_load_async_to_lds_b32 (ASYNCcnt) and read back as ds_load_b128.
// Segment softmax / aggregation -> 3-pass atomics (max, exp+sum, weighted add).
// ---------------------------------------------------------------------------

typedef __attribute__((ext_vector_type(16))) __bf16 v16bf;
typedef __attribute__((ext_vector_type(8)))  float  v8f;

#define IN_DIM  384
#define HIDDEN  256
#define KQV3    768
#define OUTD    128

// pack two f32 -> two bf16 in one dword: round (+0x8000) then v_perm_b32
__device__ __forceinline__ unsigned pack2_bf16(float lo, float hi) {
  unsigned ul = __float_as_uint(lo) + 0x8000u;
  unsigned uh = __float_as_uint(hi) + 0x8000u;
  return __builtin_amdgcn_perm(uh, ul, 0x07060302u); // {uh[31:16], ul[31:16]}
}

__device__ __forceinline__ float gelu_exact(float x) {
  return 0.5f * x * (1.0f + erff(x * 0.70710678118654752f));
}

__device__ __forceinline__ void atomicMaxFloat(float* addr, float val) {
  if (val >= 0.0f) atomicMax((int*)addr, __float_as_int(val));
  else             atomicMin((unsigned int*)addr, __float_as_uint(val));
}

union Frag { v16bf v; unsigned u[8]; };

__device__ __forceinline__ Frag pack_frag(float4 a0, float4 a1, float4 a2, float4 a3) {
  Frag f;
  f.u[0] = pack2_bf16(a0.x, a0.y);  f.u[1] = pack2_bf16(a0.z, a0.w);
  f.u[2] = pack2_bf16(a1.x, a1.y);  f.u[3] = pack2_bf16(a1.z, a1.w);
  f.u[4] = pack2_bf16(a2.x, a2.y);  f.u[5] = pack2_bf16(a2.z, a2.w);
  f.u[6] = pack2_bf16(a3.x, a3.y);  f.u[7] = pack2_bf16(a3.z, a3.w);
  return f;
}

// ---------------------------------------------------------------------------
// C[M,N] = op(A[M,K]) * B[K,N] + bias.  Block = 256 thr (8 waves).
// Block tile: 128 rows x 16 cols; wave w -> rows m0 = 128*by + 16*w.
// Per K-step: B 32x16 slab copied global->LDS (transposed [n][k], f32) via
// async-to-LDS ops shared by the block; each wave does one WMMA.
// All loads are clamped (never predicated); only C stores are guarded.
// ---------------------------------------------------------------------------
template <bool GELU_A>
__global__ void gemm_wmma_bf16(const float* __restrict__ A, int lda,
                               const float* __restrict__ B, int ldb,
                               const float* __restrict__ bias,
                               float* __restrict__ C, int ldc,
                               int M, int N, int K) {
  __shared__ float BsF[16][36];                  // [n][k], 144B rows (16B aligned)

  const int tid  = threadIdx.x;
  const int lane = tid & 31;
  const int wave = tid >> 5;
  const int r    = lane & 15;
  const int half = lane >> 4;

  const int n0 = blockIdx.x << 4;
  const int m0 = (blockIdx.y << 7) + (wave << 4);

  const int  rowA  = m0 + r;
  const int  rowCl = rowA < M ? rowA : (M - 1);  // clamp: loads always legal
  const int  colC  = n0 + r;
  const bool colOK = colC < N;
  const float* Arow = A + (long long)rowCl * lda;

  // staging coords: thread async-copies B[k0+kS][nCl] and B[k0+kS+16][nCl]
  const int nS  = tid & 15;
  const int kS  = tid >> 4;
  const int nCl = (n0 + nS) < N ? (n0 + nS) : (N - 1);
  unsigned lds0 = (unsigned)(uintptr_t)&BsF[nS][kS];
  unsigned lds1 = (unsigned)(uintptr_t)&BsF[nS][kS + 16];

  v8f acc = {};
  for (int k0 = 0; k0 < K; k0 += 32) {
    // ---- async global->LDS copy of the B slab (transpose via per-lane dst) ----
    {
      const float* g0 = B + (long long)(k0 + kS) * ldb + nCl;
      const float* g1 = B + (long long)(k0 + kS + 16) * ldb + nCl;
      asm volatile("global_load_async_to_lds_b32 %0, %1, off"
                   :: "v"(lds0), "v"(g0) : "memory");
      asm volatile("global_load_async_to_lds_b32 %0, %1, off"
                   :: "v"(lds1), "v"(g1) : "memory");
      asm volatile("s_wait_asynccnt 0x0" ::: "memory");
    }
    __syncthreads();

    // ---- A fragment: 4x unconditional global_load_b128 ----
    if (k0 + 32 < K) __builtin_prefetch(Arow + k0 + 32, 0, 0);
    float4 a0 = *(const float4*)(Arow + k0 + half * 8);
    float4 a1 = *(const float4*)(Arow + k0 + half * 8 + 4);
    float4 a2 = *(const float4*)(Arow + k0 + 16 + half * 8);
    float4 a3 = *(const float4*)(Arow + k0 + 16 + half * 8 + 4);
    if (GELU_A) {
      a0.x = gelu_exact(a0.x); a0.y = gelu_exact(a0.y); a0.z = gelu_exact(a0.z); a0.w = gelu_exact(a0.w);
      a1.x = gelu_exact(a1.x); a1.y = gelu_exact(a1.y); a1.z = gelu_exact(a1.z); a1.w = gelu_exact(a1.w);
      a2.x = gelu_exact(a2.x); a2.y = gelu_exact(a2.y); a2.z = gelu_exact(a2.z); a2.w = gelu_exact(a2.w);
      a3.x = gelu_exact(a3.x); a3.y = gelu_exact(a3.y); a3.z = gelu_exact(a3.z); a3.w = gelu_exact(a3.w);
    }
    Frag af = pack_frag(a0, a1, a2, a3);

    // ---- B fragment: col r, rows half*8..+7 and 16+half*8..+7 -> 4x ds_load_b128 + pack ----
    const float4* brow = (const float4*)&BsF[r][half * 8];
    const float4* brow2 = (const float4*)&BsF[r][16 + half * 8];
    Frag bf = pack_frag(brow[0], brow[1], brow2[0], brow2[1]);

    acc = __builtin_amdgcn_wmma_f32_16x16x32_bf16(false, af.v, false, bf.v,
                                                  (short)0, acc, false, false);
    __syncthreads();   // protect BsF before next stage
  }

  const float bv = bias ? bias[colOK ? colC : (N - 1)] : 0.0f;
#pragma unroll
  for (int rr = 0; rr < 8; ++rr) {
    int m = m0 + rr + half * 8;     // C/D layout: VGPR rr -> M=rr / rr+8
    if (m < M && colOK) C[(long long)m * ldc + colC] = acc[rr] + bv;
  }
}

__global__ void fill_f32(float* __restrict__ p, float v, long long n) {
  long long t = (long long)blockIdx.x * blockDim.x + threadIdx.x;
  if (t < n) p[t] = v;
}

// alpha_eh = (q[dst] . k_rel[src]) * p_rel[h] / sqrt(64); amax[dst,h] = max(...)
__global__ void edge_alpha_kernel(const float* __restrict__ kqv_dst, // [Nd,768]
                                  const float* __restrict__ krel,    // [Ns,256]
                                  const int* __restrict__ src,
                                  const int* __restrict__ dst,
                                  const float* __restrict__ p_rel,   // [4]
                                  float* __restrict__ alphaE,        // [E,4]
                                  float* __restrict__ amax,          // [Nd,4]
                                  int E) {
  int t = blockIdx.x * blockDim.x + threadIdx.x;
  if (t >= E * 4) return;
  int e = t >> 2, h = t & 3;
  int s = src[e], d = dst[e];
  const float4* q4 = (const float4*)(kqv_dst + (long long)d * KQV3 + HIDDEN + h * 64);
  const float4* k4 = (const float4*)(krel    + (long long)s * HIDDEN + h * 64);
  float acc = 0.0f;
#pragma unroll
  for (int i = 0; i < 16; ++i) {
    float4 qa = q4[i], ka = k4[i];
    acc += qa.x * ka.x + qa.y * ka.y + qa.z * ka.z + qa.w * ka.w;
  }
  float al = acc * p_rel[h] * 0.125f;   // 1/sqrt(64)
  alphaE[t] = al;
  atomicMaxFloat(&amax[d * 4 + h], al);
}

// ex = exp(alpha - amax[dst]); den[dst] += ex; alpha buffer now holds ex
__global__ void edge_exp_kernel(float* __restrict__ alphaE,
                                const int* __restrict__ dst,
                                const float* __restrict__ amax,
                                float* __restrict__ den, int E) {
  int t = blockIdx.x * blockDim.x + threadIdx.x;
  if (t >= E * 4) return;
  int e = t >> 2, h = t & 3;
  int d = dst[e];
  float ex = expf(alphaE[t] - amax[d * 4 + h]);
  alphaE[t] = ex;
  atomicAdd(&den[d * 4 + h], ex);
}

// agg[dst, c] += v_rel[src, c] * ex / max(den[dst,h], 1e-16); one thread per (e,c)
__global__ void edge_agg_kernel(const float* __restrict__ alphaE,
                                const float* __restrict__ den,
                                const float* __restrict__ vrel,    // [Ns,256]
                                const int* __restrict__ src,
                                const int* __restrict__ dst,
                                float* __restrict__ agg,           // [Nd,256]
                                int E) {
  long long t = (long long)blockIdx.x * blockDim.x + threadIdx.x;
  if (t >= (long long)E * HIDDEN) return;
  int c = (int)(t & (HIDDEN - 1));
  long long e = t >> 8;
  int h = c >> 6;
  int s = src[e], d = dst[e];
  float ex   = alphaE[e * 4 + h];
  float attn = ex / fmaxf(den[d * 4 + h], 1e-16f);
  atomicAdd(&agg[(long long)d * HIDDEN + c], vrel[(long long)s * HIDDEN + c] * attn);
}

// h = sigmoid(skip)*o + (1-sigmoid(skip))*h   (in place)
__global__ void skip_kernel(float* __restrict__ h, const float* __restrict__ o,
                            const float* __restrict__ skipP, long long n) {
  long long t = (long long)blockIdx.x * blockDim.x + threadIdx.x;
  if (t >= n) return;
  float sg = 1.0f / (1.0f + expf(-skipP[0]));
  h[t] = sg * o[t] + (1.0f - sg) * h[t];
}

// out = sigmoid(alpha)*out + (1-sigmoid(alpha))*base   (in place on d_out)
__global__ void blend_kernel(float* __restrict__ out, const float* __restrict__ base,
                             const float* __restrict__ alphaP, long long n) {
  long long t = (long long)blockIdx.x * blockDim.x + threadIdx.x;
  if (t >= n) return;
  float a = 1.0f / (1.0f + expf(-alphaP[0]));
  out[t] = a * out[t] + (1.0f - a) * base[t];
}

// ---------------------------------------------------------------------------
extern "C" void kernel_launch(void* const* d_in, const int* in_sizes, int n_in,
                              void* d_out, int out_size, void* d_ws, size_t ws_size,
                              hipStream_t stream) {
  (void)n_in; (void)out_size; (void)ws_size;
  // inputs 0..12: features + edge indices
  const float* x[3] = {(const float*)d_in[0], (const float*)d_in[1], (const float*)d_in[2]};
  int N[3] = {in_sizes[0] / IN_DIM, in_sizes[1] / IN_DIM, in_sizes[2] / IN_DIM};

  const int eidx[5] = {3, 5, 7, 9, 11};     // cites, pv, vp, pf, fp
  const int* esrc[5]; const int* edst[5]; int E[5];
  for (int r = 0; r < 5; ++r) {
    esrc[r] = (const int*)d_in[eidx[r]];
    edst[r] = (const int*)d_in[eidx[r] + 1];
    E[r]    = in_sizes[eidx[r]];
  }
  const int srcT[5] = {0, 0, 1, 0, 2};      // paper=0, venue=1, field=2
  const int dstT[5] = {0, 1, 0, 2, 0};
  // position of each relation within alphabetically-sorted {cites,fp,pf,pv,vp}
  const int relOff[5] = {0, 3, 4, 2, 1};

  // params: alphabetical pytree flattening starting at d_in[13]
  // 0: alpha | 1..30: conv0 | 31..60: conv1 | 61..66: lin | 67,68: out | 69,70: paper_proj
  auto L = [&](int i) { return (const float*)d_in[13 + i]; };
  const float* p_alpha = L(0);
  const int linW[3] = {63, 65, 61}, linB[3] = {64, 66, 62};       // types sorted f,p,v
  const float* outW  = L(67); const float* outB  = L(68);
  const float* projW = L(69); const float* projB = L(70);
  // within a conv layer (base b0): a_rel+0..4, kqv+5..10, m_rel+11..15,
  // out_lin+16..21, p_rel+22..26, skip+27..29
  const int kqvW[3]  = {7, 9, 5},   kqvB[3]  = {8, 10, 6};
  const int olW[3]   = {18, 20, 16}, olB[3]  = {19, 21, 17};
  const int skipO[3] = {28, 29, 27};

  // ---- workspace carve-up (f32) ----
  float* w = (float*)d_ws;
  size_t off = 0;
  auto alloc = [&](size_t n) { float* p = w + off; off += n; return p; };
  float *h[3], *kqv[3], *agg[3], *amax[3], *den[3];
  for (int t = 0; t < 3; ++t) h[t]    = alloc((size_t)N[t] * HIDDEN);
  for (int t = 0; t < 3; ++t) kqv[t]  = alloc((size_t)N[t] * KQV3);
  for (int t = 0; t < 3; ++t) agg[t]  = alloc((size_t)N[t] * HIDDEN);
  for (int t = 0; t < 3; ++t) amax[t] = alloc((size_t)N[t] * 4);
  for (int t = 0; t < 3; ++t) den[t]  = alloc((size_t)N[t] * 4);
  int maxN = N[0];
  float* krel = alloc((size_t)maxN * HIDDEN);
  float* vrel = alloc((size_t)maxN * HIDDEN);
  float* obuf = alloc((size_t)maxN * HIDDEN);
  float* aE[5];
  for (int r = 0; r < 5; ++r) aE[r] = alloc((size_t)E[r] * 4);
  float* base = alloc((size_t)N[0] * OUTD);

  auto gemm = [&](const float* A, int lda, const float* B, int ldb,
                  const float* bias, float* C, int ldc,
                  int M, int Nn, int K, bool geluA) {
    dim3 grid((Nn + 15) >> 4, (((M + 15) >> 4) + 7) >> 3);   // (colTiles, rowTiles/8)
    if (geluA)
      gemm_wmma_bf16<true><<<grid, 256, 0, stream>>>(A, lda, B, ldb, bias, C, ldc, M, Nn, K);
    else
      gemm_wmma_bf16<false><<<grid, 256, 0, stream>>>(A, lda, B, ldb, bias, C, ldc, M, Nn, K);
  };
  auto fill = [&](float* p, float v, long long n) {
    fill_f32<<<(int)((n + 255) / 256), 256, 0, stream>>>(p, v, n);
  };

  // ---- input linear per node type ----
  for (int t = 0; t < 3; ++t)
    gemm(x[t], IN_DIM, L(linW[t]), HIDDEN, L(linB[t]), h[t], HIDDEN, N[t], HIDDEN, IN_DIM, false);

  // ---- 2 HGT conv layers ----
  for (int l = 0; l < 2; ++l) {
    int b0 = 1 + l * 30;
    for (int t = 0; t < 3; ++t)
      gemm(h[t], HIDDEN, L(b0 + kqvW[t]), KQV3, L(b0 + kqvB[t]), kqv[t], KQV3, N[t], KQV3, HIDDEN, false);
    for (int t = 0; t < 3; ++t) {
      fill(amax[t], -3.0e38f, (long long)N[t] * 4);
      fill(den[t], 0.0f, (long long)N[t] * 4);
      fill(agg[t], 0.0f, (long long)N[t] * HIDDEN);
    }
    // pass A: per-relation k_rel (4 per-head 64x64 GEMMs) + alpha + scatter-max
    for (int r = 0; r < 5; ++r) {
      const float* arel = L(b0 + relOff[r]);
      for (int hh = 0; hh < 4; ++hh)
        gemm(kqv[srcT[r]] + hh * 64, KQV3, arel + hh * 4096, 64, nullptr,
             krel + hh * 64, HIDDEN, N[srcT[r]], 64, 64, false);
      int thr = E[r] * 4;
      edge_alpha_kernel<<<(thr + 255) / 256, 256, 0, stream>>>(
          kqv[dstT[r]], krel, esrc[r], edst[r], L(b0 + 22 + relOff[r]),
          aE[r], amax[dstT[r]], E[r]);
    }
    // pass B: exp + scatter-sum of denominators (amax final across relations)
    for (int r = 0; r < 5; ++r) {
      int thr = E[r] * 4;
      edge_exp_kernel<<<(thr + 255) / 256, 256, 0, stream>>>(
          aE[r], edst[r], amax[dstT[r]], den[dstT[r]], E[r]);
    }
    // pass C: per-relation v_rel + attention-weighted scatter-sum
    for (int r = 0; r < 5; ++r) {
      const float* mrel = L(b0 + 11 + relOff[r]);
      for (int hh = 0; hh < 4; ++hh)
        gemm(kqv[srcT[r]] + 512 + hh * 64, KQV3, mrel + hh * 4096, 64, nullptr,
             vrel + hh * 64, HIDDEN, N[srcT[r]], 64, 64, false);
      long long thr = (long long)E[r] * HIDDEN;
      edge_agg_kernel<<<(int)((thr + 255) / 256), 256, 0, stream>>>(
          aE[r], den[dstT[r]], vrel, esrc[r], edst[r], agg[dstT[r]], E[r]);
    }
    // gelu + out_lin (fused) + sigmoid-skip residual
    for (int t = 0; t < 3; ++t) {
      gemm(agg[t], HIDDEN, L(b0 + olW[t]), HIDDEN, L(b0 + olB[t]),
           obuf, HIDDEN, N[t], HIDDEN, HIDDEN, true);
      long long n = (long long)N[t] * HIDDEN;
      skip_kernel<<<(int)((n + 255) / 256), 256, 0, stream>>>(
          h[t], obuf, L(b0 + skipO[t]), n);
    }
  }

  // ---- head: gnn out + paper_proj + sigmoid blend ----
  gemm(h[0], HIDDEN, outW, OUTD, outB, (float*)d_out, OUTD, N[0], OUTD, HIDDEN, false);
  gemm(x[0], IN_DIM, projW, OUTD, projB, base, OUTD, N[0], OUTD, IN_DIM, false);
  long long n = (long long)N[0] * OUTD;
  blend_kernel<<<(int)((n + 255) / 256), 256, 0, stream>>>((float*)d_out, base, p_alpha, n);
}
